// Attention_34127810134504
// MI455X (gfx1250) — matrix operand order repaired and checked by reading, not compiled, not run on hardware
//
#include <hip/hip_runtime.h>
#include <math.h>

typedef __attribute__((ext_vector_type(16))) __bf16 v16bf;
typedef __attribute__((ext_vector_type(8)))  float  v8f;

#define H_   8
#define DH_  64
#define DIM_ 512
#define Q_   16
#define N_   512
#define TOK_ 8192          // Q_*N_
#define NROW 65536         // H_*Q_*N_
#define CNT_ 33554432.0    // H*Q*N*M

// -------- workspace layout (bytes) --------
#define OFF_FQ    0u
#define OFF_FK    8388608u
#define OFF_FV    16777216u
#define OFF_FVT   25165824u
#define OFF_WINT  33554432u
#define OFF_WOUTT 34078720u
#define OFF_HOUT  34603008u
#define OFF_NQ    42991616u
#define OFF_NK    43253760u
#define OFF_MUQ   43515904u
#define OFF_SK    43778048u
#define OFF_A2    44040192u
#define OFF_MUK   44302336u
#define OFF_QG    44335104u
#define OFF_KG    44337152u
#define OFF_MROW  44339200u
#define OFF_GSUM  44601344u
#define OFF_SCAL  44601408u

#if defined(__HIP_DEVICE_COMPILE__) && defined(__gfx1250__) && \
    __has_builtin(__builtin_amdgcn_global_load_async_to_lds_b128) && \
    __has_builtin(__builtin_amdgcn_s_wait_asynccnt)
#define HAVE_ASYNC_LDS 1
#else
#define HAVE_ASYNC_LDS 0
#endif

// cov scale: (LAMBDA/M) / (sqrt(DH)+1e-8)
__device__ __constant__ float SCOV = 2.4414062e-07f;

// ---------------- helpers ----------------
__device__ inline float wave_sum(float v) {
  #pragma unroll
  for (int o = 16; o > 0; o >>= 1) v += __shfl_down(v, o, 32);
  return v;
}
__device__ inline float wave_max(float v) {
  #pragma unroll
  for (int o = 16; o > 0; o >>= 1) v = fmaxf(v, __shfl_down(v, o, 32));
  return v;
}

// A fragment (16x32 bf16, row-major, lane holds row lane&15, g = lane>>4)
__device__ inline v16bf frag_a(const __bf16* rowptr, int k0, int g) {
  v16bf f;
  const __bf16* p = rowptr + k0 + 8 * g;
  #pragma unroll
  for (int j = 0; j < 8; ++j) f[j] = p[j];
  #pragma unroll
  for (int j = 0; j < 8; ++j) f[j + 8] = p[16 + j];
  return f;
}
// B fragment (32x16 bf16) loaded from B^T rows (K contiguous); K = 8*g + j, 16 + 8*g + j
__device__ inline v16bf frag_b(const __bf16* colptr, int k0, int g) {
  v16bf f;
  const __bf16* p = colptr + k0 + 8 * g;
  #pragma unroll
  for (int j = 0; j < 8; ++j) f[j] = p[j];
  #pragma unroll
  for (int j = 0; j < 8; ++j) f[j + 8] = p[16 + j];
  return f;
}
__device__ inline v8f wmma_bf16(v16bf a, v16bf b, v8f c) {
  return __builtin_amdgcn_wmma_f32_16x16x32_bf16(false, a, false, b, (short)0, c,
                                                 false, false);
}
__device__ inline float std1f(float s, float s2, double cnt) {
  double var = ((double)s2 - (double)s * (double)s / cnt) / (cnt - 1.0);
  return (float)sqrt(var > 0.0 ? var : 0.0);
}

// ---------------- kernels ----------------
__global__ void k_zero(float* mrow, float* gsum) {
  int i = blockIdx.x * blockDim.x + threadIdx.x;
  if (i < NROW) mrow[i] = 0.f;
  if (i < 8) gsum[i] = 0.f;
}

__global__ void k_prep(const float* __restrict__ Win, const float* __restrict__ Wout,
                       __bf16* __restrict__ Wint, __bf16* __restrict__ Woutt) {
  int i = blockIdx.x * blockDim.x + threadIdx.x;   // 262144
  int r = i >> 9, c = i & 511;
  Wint[(size_t)c * 512 + r]  = (__bf16)Win[i];
  Woutt[(size_t)c * 512 + r] = (__bf16)Wout[i];
}

// LayerNorm + GEMM (16 tokens per block, 4 waves)
__global__ __launch_bounds__(128) void k_proj(const float* __restrict__ x,
    const float* __restrict__ gamma, const float* __restrict__ beta,
    const __bf16* __restrict__ Wt, __bf16* __restrict__ out) {
  __shared__ __bf16 A[16 * 512];
  int t0 = blockIdx.x * 16;
  int lane = threadIdx.x & 31, wave = threadIdx.x >> 5;
  #pragma unroll
  for (int i = 0; i < 4; ++i) {
    int row = wave * 4 + i;
    const float* xr = x + (size_t)(t0 + row) * 512;
    float s = 0.f, s2 = 0.f;
    #pragma unroll
    for (int j = 0; j < 16; ++j) { float v = xr[lane + 32 * j]; s += v; s2 += v * v; }
    s = wave_sum(s); s2 = wave_sum(s2);
    s = __shfl(s, 0, 32); s2 = __shfl(s2, 0, 32);
    float mu = s * (1.f / 512.f);
    float rstd = rsqrtf(s2 * (1.f / 512.f) - mu * mu + 1e-5f);
    #pragma unroll
    for (int j = 0; j < 16; ++j) {
      int c = lane + 32 * j;
      A[row * 512 + c] = (__bf16)((xr[c] - mu) * rstd * gamma[c] + beta[c]);
    }
  }
  __syncthreads();
  int g = lane >> 4, lm = lane & 15;
  const __bf16* arow = A + (size_t)lm * 512;
  for (int ct = wave; ct < 32; ct += 4) {
    int col0 = ct * 16;
    const __bf16* brow = Wt + (size_t)(col0 + lm) * 512;
    v8f acc = {};
    for (int k0 = 0; k0 < 512; k0 += 32) {
      if (k0 + 64 < 512) __builtin_prefetch(brow + k0 + 64, 0, 3);
      acc = wmma_bf16(frag_a(arow, k0, g), frag_b(brow, k0, g), acc);
    }
    int c = col0 + lm;
    #pragma unroll
    for (int r = 0; r < 8; ++r)
      out[(size_t)(t0 + r + 8 * g) * 512 + c] = (__bf16)acc[r];
  }
}

__global__ void k_fvt(const __bf16* __restrict__ fv, __bf16* __restrict__ fvt) {
  int i = blockIdx.x * blockDim.x + threadIdx.x;   // 4194304
  int m = i & 511, rest = i >> 9, d = rest & 63, hq = rest >> 6;
  int h = hq >> 4, q = hq & 15;
  fvt[i] = fv[((size_t)(q * 512 + m)) * 512 + h * 64 + d];
}

__global__ void k_colmean(const __bf16* __restrict__ fk, float* __restrict__ muk) {
  int i = blockIdx.x * blockDim.x + threadIdx.x;   // 8192 = hq*64+d
  int d = i & 63, hq = i >> 6, h = hq >> 4, q = hq & 15;
  const __bf16* p = fk + ((size_t)q * 512) * 512 + h * 64 + d;
  float s = 0.f;
  for (int m = 0; m < 512; ++m) s += (float)p[(size_t)m * 512];
  muk[i] = s * (1.f / 512.f);
}

__global__ void k_rowstats_q(const __bf16* __restrict__ fq, const float* __restrict__ muk,
                             float* __restrict__ nq, float* __restrict__ muq,
                             float* __restrict__ a2) {
  int rid = blockIdx.x * blockDim.x + threadIdx.x;  // 65536
  int n = rid & 511, hq = rid >> 9, h = hq >> 4, q = hq & 15;
  const __bf16* p = fq + ((size_t)(q * 512 + n)) * 512 + h * 64;
  const float* mk = muk + hq * 64;
  float s = 0.f, s2 = 0.f, dm = 0.f, sm = 0.f;
  for (int d = 0; d < 64; ++d) {
    float v = (float)p[d], u = mk[d];
    s += v; s2 += v * v; dm += v * u; sm += u;
  }
  nq[rid] = sqrtf(s2);
  float mu = s * (1.f / 64.f);
  muq[rid] = mu;
  a2[rid] = dm - mu * sm;
}

__global__ void k_rowstats_k(const __bf16* __restrict__ fk,
                             float* __restrict__ nk, float* __restrict__ sk) {
  int rid = blockIdx.x * blockDim.x + threadIdx.x;
  int m = rid & 511, hq = rid >> 9, h = hq >> 4, q = hq & 15;
  const __bf16* p = fk + ((size_t)(q * 512 + m)) * 512 + h * 64;
  float s = 0.f, s2 = 0.f;
  for (int d = 0; d < 64; ++d) { float v = (float)p[d]; s += v; s2 += v * v; }
  nk[rid] = sqrtf(s2);
  sk[rid] = s;
}

__global__ void k_qgkg(const __bf16* __restrict__ fq, const __bf16* __restrict__ fk,
                       float* __restrict__ qg, float* __restrict__ kg) {
  int i = blockIdx.x * blockDim.x + threadIdx.x;   // 1024
  int sel = i >> 9, hd = i & 511, h = hd >> 6, d = hd & 63;
  const __bf16* src = (sel ? fk : fq) + h * 64 + d;
  float s = 0.f;
  for (int t = 0; t < TOK_; ++t) s += (float)src[(size_t)t * 512];
  (sel ? kg : qg)[hd] = s * (1.f / (float)TOK_);
}

// WMMA score recompute pass. mode 0: cos/cov stats + margin row sums.
// mode 1: dots stats with per-head coefficients.
__global__ __launch_bounds__(128) void k_score_pass(
    const __bf16* __restrict__ fq, const __bf16* __restrict__ fk,
    const float* __restrict__ nq, const float* __restrict__ nk,
    const float* __restrict__ muq, const float* __restrict__ sk,
    const float* __restrict__ a2, float* __restrict__ mrow,
    float* __restrict__ gsum, const float* __restrict__ scal, int mode) {
  int bid = blockIdx.x;
  int ntile = bid & 31, hq = bid >> 5, h = hq >> 4, q = hq & 15;
  int lane = threadIdx.x & 31, wave = threadIdx.x >> 5;
  int g = lane >> 4, lm = lane & 15;
  int n0 = ntile * 16, rbase = hq * 512 + n0;
  const __bf16* qbase = fq + ((size_t)q * 512) * 512 + h * 64;
  const __bf16* kbase = fk + ((size_t)q * 512) * 512 + h * 64;
  const __bf16* arow = qbase + (size_t)(n0 + lm) * 512;
  // A fragments are invariant across m-tiles: hoist out of the loop
  v16bf a0 = frag_a(arow, 0, g);
  v16bf a1 = frag_a(arow, 32, g);
  float nq_r[8], muq_r[8], a2_r[8], v_r[8], marg[8];
  #pragma unroll
  for (int r = 0; r < 8; ++r) {
    int rid = rbase + r + 8 * g;
    nq_r[r] = nq[rid]; muq_r[r] = muq[rid]; a2_r[r] = a2[rid];
    v_r[r] = mode ? mrow[rid] * (1.f / 512.f) : 0.f;
    marg[r] = 0.f;
  }
  float cA = 0.f, cB = 0.f, cC = 0.f;
  if (mode) { cA = scal[h]; cB = scal[8 + h]; cC = scal[16 + h]; }
  float s0 = 0.f, s1 = 0.f, s2 = 0.f, s3 = 0.f;
  for (int mt = wave; mt < 32; mt += 4) {
    const __bf16* brow = kbase + (size_t)(mt * 16 + lm) * 512;
    v8f acc = {};
    acc = wmma_bf16(a0, frag_b(brow, 0, g),  acc);
    acc = wmma_bf16(a1, frag_b(brow, 32, g), acc);
    int mcol = hq * 512 + mt * 16 + lm;
    float nk_m = nk[mcol], sk_m = sk[mcol];
    #pragma unroll
    for (int r = 0; r < 8; ++r) {
      float dot = acc[r];
      float cosv = dot / ((nq_r[r] + 1e-8f) * (nk_m + 1e-8f));
      cosv = fminf(fmaxf(cosv, -0.99f), 0.99f);
      float cov = SCOV * (dot - a2_r[r] - muq_r[r] * sk_m);
      cov = fminf(fmaxf(cov, -10.f), 10.f);
      if (mode == 0) {
        float csv = dot / (fmaxf(nq_r[r], 1e-8f) * fmaxf(nk_m, 1e-8f));
        csv = fminf(fmaxf(csv, -0.99f), 0.99f);
        s0 += cosv; s1 += cosv * cosv; s2 += cov; s3 += cov * cov;
        marg[r] += fminf(fmaxf(0.01f - csv, 0.f), 2.f);
      } else {
        float dv = cA * cosv + cB * cov + cC * v_r[r];
        s0 += dv; s1 += dv * dv;
      }
    }
  }
  if (mode == 0) {
    #pragma unroll
    for (int r = 0; r < 8; ++r) atomicAdd(&mrow[rbase + r + 8 * g], marg[r]);
    s0 = wave_sum(s0); s1 = wave_sum(s1); s2 = wave_sum(s2); s3 = wave_sum(s3);
    if (lane == 0) {
      atomicAdd(&gsum[0], s0); atomicAdd(&gsum[1], s1);
      atomicAdd(&gsum[2], s2); atomicAdd(&gsum[3], s3);
    }
  } else {
    s0 = wave_sum(s0); s1 = wave_sum(s1);
    if (lane == 0) { atomicAdd(&gsum[4], s0); atomicAdd(&gsum[5], s1); }
  }
}

// per-head MLP + sigma(cos/cov/var) -> mixing coefficients
__global__ __launch_bounds__(32) void k_mlp_sigma(
    const float* __restrict__ qg, const float* __restrict__ kg,
    const float* __restrict__ w1, const float* __restrict__ b1,
    const float* __restrict__ lg, const float* __restrict__ lb,
    const float* __restrict__ w2, const float* __restrict__ b2,
    const float* __restrict__ w3, const float* __restrict__ b3,
    const float* __restrict__ temp, const float* __restrict__ mrow,
    const float* __restrict__ gsum, float* __restrict__ scal) {
  int lane = threadIdx.x;
  float vs = 0.f, vs2 = 0.f;
  for (int i = lane; i < NROW; i += 32) {
    float v = mrow[i] * (1.f / 512.f);
    vs += v; vs2 += v * v;
  }
  vs = wave_sum(vs); vs2 = wave_sum(vs2);
  vs = __shfl(vs, 0, 32) * 512.f;     // broadcast over M multiplicity
  vs2 = __shfl(vs2, 0, 32) * 512.f;
  float sig_var = std1f(vs, vs2, CNT_);
  float sig_cos = std1f(gsum[0], gsum[1], CNT_);
  float sig_cov = std1f(gsum[2], gsum[3], CNT_);
  if (lane < 8) {
    float feats[128];
    for (int i = 0; i < 64; ++i) { feats[i] = qg[lane * 64 + i]; feats[64 + i] = kg[lane * 64 + i]; }
    float h1[64];
    for (int j = 0; j < 64; ++j) {
      float a = b1[j];
      for (int i = 0; i < 128; ++i) a += feats[i] * w1[i * 64 + j];
      h1[j] = a;
    }
    float mu = 0.f, var = 0.f;
    for (int j = 0; j < 64; ++j) mu += h1[j];
    mu *= (1.f / 64.f);
    for (int j = 0; j < 64; ++j) { float d = h1[j] - mu; var += d * d; }
    var *= (1.f / 64.f);
    float rstd = rsqrtf(var + 1e-5f);
    for (int j = 0; j < 64; ++j)
      h1[j] = fmaxf((h1[j] - mu) * rstd * lg[j] + lb[j], 0.f);
    float h2[32];
    for (int j = 0; j < 32; ++j) {
      float a = b2[j];
      for (int i = 0; i < 64; ++i) a += h1[i] * w2[i * 32 + j];
      h2[j] = fmaxf(a, 0.f);
    }
    float l3[3];
    for (int j = 0; j < 3; ++j) {
      float a = b3[j];
      for (int i = 0; i < 32; ++i) a += h2[i] * w3[i * 3 + j];
      l3[j] = a;
    }
    float mx = fmaxf(l3[0], fmaxf(l3[1], l3[2])), es = 0.f;
    for (int j = 0; j < 3; ++j) { l3[j] = __expf(l3[j] - mx); es += l3[j]; }
    float t = fminf(fmaxf(temp[0], 0.1f), 20.f);
    float w[3]; float es2 = 0.f;
    float mx2 = -1e30f;
    for (int j = 0; j < 3; ++j) { w[j] = (l3[j] / es) / t; mx2 = fmaxf(mx2, w[j]); }
    for (int j = 0; j < 3; ++j) { w[j] = __expf(w[j] - mx2); es2 += w[j]; }
    scal[lane]      = (w[0] / es2) / (sig_cos + 1e-8f);
    scal[8 + lane]  = (w[1] / es2) / (sig_cov + 1e-8f) * 0.1f;
    scal[16 + lane] = (w[2] / es2) / (sig_var + 1e-8f) * 0.1f;
  }
}

__global__ void k_invt(const float* __restrict__ gsum, float* __restrict__ scal) {
  float sig = std1f(gsum[4], gsum[5], CNT_);
  float t = fminf(fmaxf(1.f + 0.5f * sig, 0.5f), 5.f);
  scal[24] = 1.f / t;
}

// fused dots -> softmax -> P @ V per (h,q,16-row tile)
__global__ __launch_bounds__(128) void k_softmax_av(
    const __bf16* __restrict__ fq, const __bf16* __restrict__ fk,
    const __bf16* __restrict__ fvt,
    const float* __restrict__ nq, const float* __restrict__ nk,
    const float* __restrict__ muq, const float* __restrict__ sk,
    const float* __restrict__ a2, const float* __restrict__ mrow,
    const float* __restrict__ scal, __bf16* __restrict__ hout) {
  __shared__ float S[16 * 512];
  __shared__ __bf16 P[16 * 512];
  __shared__ float rsum[16];
  int bid = blockIdx.x;
  int ntile = bid & 31, hq = bid >> 5, h = hq >> 4, q = hq & 15;
  int lane = threadIdx.x & 31, wave = threadIdx.x >> 5;
  int g = lane >> 4, lm = lane & 15;
  int n0 = ntile * 16, rbase = hq * 512 + n0;
  const __bf16* qbase = fq + ((size_t)q * 512) * 512 + h * 64;
  const __bf16* kbase = fk + ((size_t)q * 512) * 512 + h * 64;
  const __bf16* arow = qbase + (size_t)(n0 + lm) * 512;
  v16bf a0 = frag_a(arow, 0, g);
  v16bf a1 = frag_a(arow, 32, g);
  float nq_r[8], muq_r[8], a2_r[8], v_r[8];
  #pragma unroll
  for (int r = 0; r < 8; ++r) {
    int rid = rbase + r + 8 * g;
    nq_r[r] = nq[rid]; muq_r[r] = muq[rid]; a2_r[r] = a2[rid];
    v_r[r] = mrow[rid] * (1.f / 512.f);
  }
  float cA = scal[h], cB = scal[8 + h], cC = scal[16 + h], invt = scal[24];
  for (int mt = wave; mt < 32; mt += 4) {
    const __bf16* brow = kbase + (size_t)(mt * 16 + lm) * 512;
    v8f acc = {};
    acc = wmma_bf16(a0, frag_b(brow, 0, g),  acc);
    acc = wmma_bf16(a1, frag_b(brow, 32, g), acc);
    int mcol = hq * 512 + mt * 16 + lm;
    float nk_m = nk[mcol], sk_m = sk[mcol];
    #pragma unroll
    for (int r = 0; r < 8; ++r) {
      float dot = acc[r];
      float cosv = dot / ((nq_r[r] + 1e-8f) * (nk_m + 1e-8f));
      cosv = fminf(fmaxf(cosv, -0.99f), 0.99f);
      float cov = SCOV * (dot - a2_r[r] - muq_r[r] * sk_m);
      cov = fminf(fmaxf(cov, -10.f), 10.f);
      S[(r + 8 * g) * 512 + mt * 16 + lm] = (cA * cosv + cB * cov + cC * v_r[r]) * invt;
    }
  }
  __syncthreads();
  #pragma unroll
  for (int rr = 0; rr < 4; ++rr) {
    int row = wave * 4 + rr;
    float mx = -1e30f;
    for (int j = lane; j < 512; j += 32) mx = fmaxf(mx, S[row * 512 + j]);
    mx = wave_max(mx); mx = __shfl(mx, 0, 32);
    float s = 0.f;
    for (int j = lane; j < 512; j += 32) {
      float e = __expf(S[row * 512 + j] - mx);
      P[row * 512 + j] = (__bf16)e;
      s += e;
    }
    s = wave_sum(s);
    if (lane == 0) rsum[row] = s;
  }
  __syncthreads();
  int d0 = wave * 16;
  const __bf16* prow = P + (size_t)lm * 512;
  const __bf16* brow = fvt + ((size_t)hq * 64 + d0 + lm) * 512;
  v8f acc = {};
  for (int k0 = 0; k0 < 512; k0 += 32) {
    if (k0 + 64 < 512) __builtin_prefetch(brow + k0 + 64, 0, 3);
    acc = wmma_bf16(frag_a(prow, k0, g), frag_b(brow, k0, g), acc);
  }
  int c = h * 64 + d0 + lm;
  #pragma unroll
  for (int r = 0; r < 8; ++r) {
    int n = n0 + r + 8 * g;
    hout[(size_t)(q * 512 + n) * 512 + c] = (__bf16)(acc[r] / rsum[r + 8 * g]);
  }
}

__global__ __launch_bounds__(128) void k_outproj(
    const __bf16* __restrict__ hout, const __bf16* __restrict__ Wt,
    const float* __restrict__ bout, float* __restrict__ out) {
  __shared__ __bf16 A[16 * 512];
  int t0 = blockIdx.x * 16;
#if HAVE_ASYNC_LDS
  {
    // gfx1250 async global->LDS copy: 16KB tile as 1024 x b128 transfers.
    // builtin expects int4 (vector_size(16)) pointers in addrspace(1)/(3).
    typedef int v4i __attribute__((vector_size(4 * sizeof(int))));
    typedef __attribute__((address_space(1))) v4i gv4;
    typedef __attribute__((address_space(3))) v4i lv4;
    gv4* gsrc = (gv4*)(hout + (size_t)t0 * 512);
    lv4* ldst = (lv4*)A;
    for (int i = threadIdx.x; i < 1024; i += 128)
      __builtin_amdgcn_global_load_async_to_lds_b128(gsrc + i, ldst + i, 0, 0);
    __builtin_amdgcn_s_wait_asynccnt(0);
  }
#else
  for (int i = threadIdx.x; i < 16 * 512; i += 128) A[i] = hout[(size_t)t0 * 512 + i];
#endif
  __syncthreads();
  int lane = threadIdx.x & 31, wave = threadIdx.x >> 5;
  int g = lane >> 4, lm = lane & 15;
  const __bf16* arow = A + (size_t)lm * 512;
  for (int ct = wave; ct < 32; ct += 4) {
    int col0 = ct * 16;
    const __bf16* brow = Wt + (size_t)(col0 + lm) * 512;
    v8f acc = {};
    for (int k0 = 0; k0 < 512; k0 += 32) {
      if (k0 + 64 < 512) __builtin_prefetch(brow + k0 + 64, 0, 3);
      acc = wmma_bf16(frag_a(arow, k0, g), frag_b(brow, k0, g), acc);
    }
    int c = col0 + lm;
    float bb = bout[c];
    #pragma unroll
    for (int r = 0; r < 8; ++r)
      out[(size_t)(t0 + r + 8 * g) * 512 + c] = acc[r] + bb;
  }
}

// ---------------- host launcher ----------------
extern "C" void kernel_launch(void* const* d_in, const int* in_sizes, int n_in,
                              void* d_out, int out_size, void* d_ws, size_t ws_size,
                              hipStream_t stream) {
  (void)in_sizes; (void)n_in; (void)out_size; (void)ws_size;
  const float* q_in  = (const float*)d_in[0];
  const float* k_in  = (const float*)d_in[1];
  const float* v_in  = (const float*)d_in[2];
  const float* ln_g  = (const float*)d_in[3];
  const float* ln_b  = (const float*)d_in[4];
  const float* W_in  = (const float*)d_in[5];
  const float* wp1_w = (const float*)d_in[6];
  const float* wp1_b = (const float*)d_in[7];
  const float* wpln_g= (const float*)d_in[8];
  const float* wpln_b= (const float*)d_in[9];
  const float* wp2_w = (const float*)d_in[10];
  const float* wp2_b = (const float*)d_in[11];
  const float* wp3_w = (const float*)d_in[12];
  const float* wp3_b = (const float*)d_in[13];
  const float* temp  = (const float*)d_in[14];
  const float* W_out = (const float*)d_in[15];
  const float* b_out = (const float*)d_in[16];

  char* ws = (char*)d_ws;
  __bf16* fq    = (__bf16*)(ws + OFF_FQ);
  __bf16* fk    = (__bf16*)(ws + OFF_FK);
  __bf16* fv    = (__bf16*)(ws + OFF_FV);
  __bf16* fvt   = (__bf16*)(ws + OFF_FVT);
  __bf16* wint  = (__bf16*)(ws + OFF_WINT);
  __bf16* woutt = (__bf16*)(ws + OFF_WOUTT);
  __bf16* hout  = (__bf16*)(ws + OFF_HOUT);
  float* nq   = (float*)(ws + OFF_NQ);
  float* nk   = (float*)(ws + OFF_NK);
  float* muq  = (float*)(ws + OFF_MUQ);
  float* sk   = (float*)(ws + OFF_SK);
  float* a2   = (float*)(ws + OFF_A2);
  float* muk  = (float*)(ws + OFF_MUK);
  float* qg   = (float*)(ws + OFF_QG);
  float* kg   = (float*)(ws + OFF_KG);
  float* mrow = (float*)(ws + OFF_MROW);
  float* gsum = (float*)(ws + OFF_GSUM);
  float* scal = (float*)(ws + OFF_SCAL);
  float* out  = (float*)d_out;

  k_zero<<<256, 256, 0, stream>>>(mrow, gsum);
  k_prep<<<1024, 256, 0, stream>>>(W_in, W_out, wint, woutt);
  k_proj<<<512, 128, 0, stream>>>(q_in, ln_g, ln_b, wint, fq);
  k_proj<<<512, 128, 0, stream>>>(k_in, ln_g, ln_b, wint, fk);
  k_proj<<<512, 128, 0, stream>>>(v_in, ln_g, ln_b, wint, fv);
  k_fvt<<<16384, 256, 0, stream>>>(fv, fvt);
  k_colmean<<<32, 256, 0, stream>>>(fk, muk);
  k_rowstats_q<<<256, 256, 0, stream>>>(fq, muk, nq, muq, a2);
  k_rowstats_k<<<256, 256, 0, stream>>>(fk, nk, sk);
  k_qgkg<<<4, 256, 0, stream>>>(fq, fk, qg, kg);
  k_score_pass<<<4096, 128, 0, stream>>>(fq, fk, nq, nk, muq, sk, a2, mrow, gsum,
                                         scal, 0);
  k_mlp_sigma<<<1, 32, 0, stream>>>(qg, kg, wp1_w, wp1_b, wpln_g, wpln_b,
                                    wp2_w, wp2_b, wp3_w, wp3_b, temp, mrow, gsum,
                                    scal);
  k_score_pass<<<4096, 128, 0, stream>>>(fq, fk, nq, nk, muq, sk, a2, mrow, gsum,
                                         scal, 1);
  k_invt<<<1, 1, 0, stream>>>(gsum, scal);
  k_softmax_av<<<4096, 128, 0, stream>>>(fq, fk, fvt, nq, nk, muq, sk, a2, mrow,
                                         scal, hout);
  k_outproj<<<512, 128, 0, stream>>>(hout, woutt, b_out, out);
}
